// TransformerLSModel_38680475468081
// MI455X (gfx1250) — compile-verified, matrix operand order checked by reading, not run on hardware
//
#include <hip/hip_runtime.h>
#include <hip/hip_bf16.h>
#include <math.h>

// ---------------- model constants ----------------
#define VV   8192
#define DD   1024
#define DFF  4096
#define HH   16
#define NLAY 4
#define BB   2
#define LL   2048
#define MM   2048
#define SS   64
#define WW   512
#define DHD  64
#define NCV  64          // compressed kv rows per batch
#define WLEN (WW + LL)   // 2560 rows of hw/kw/vw per batch

// attention LDS score layout: [16 queries][608 cols]
#define RECTC 544
#define SCW   608

typedef __attribute__((ext_vector_type(16))) _Float16 v16h;
typedef __attribute__((ext_vector_type(8)))  float    v8f;

union AF { v16h v; _Float16 h[16]; unsigned u[8]; };
union CF { v8f  v; float f[8]; };

__device__ inline v8f v8f_zero() {
  CF c;
#pragma unroll
  for (int i = 0; i < 8; ++i) c.f[i] = 0.f;
  return c.v;
}

// =====================================================================
// Generic f32-in / f32-out GEMM via v_wmma_f32_16x16x32_f16.
// C[M,N] = act(A[M,K] @ B + bias); transB: B stored [N,K] row-major.
// Block = 128 threads (4 waves, 2x2 wave grid), tile 128x128, BK=32.
// Each wave owns a 64x64 sub-tile = 16 WMMA accumulators -> 1.0 LDS
// b128 load per WMMA. K%32==0; M,N arbitrary (guarded).
// =====================================================================
#define BM 128
#define BN 128
#define BK 32
#define LDP (BK + 8)   // padded LDS row stride: 40 halfwords = 80B (16B-mult)

__global__ __launch_bounds__(128) void k_gemm(
    const float* __restrict__ A, const float* __restrict__ B,
    const float* __restrict__ bias, float* __restrict__ C,
    int M, int N, int K, int transB, int act)
{
  __shared__ _Float16 As[BM][LDP];
  __shared__ _Float16 Bs[BN][LDP];   // stored [n][k]

  const int tid  = threadIdx.x;
  const int lane = tid & 31;
  const int wave = tid >> 5;
  const int hi   = lane >> 4;
  const int ln   = lane & 15;
  const int wr   = (wave & 1) * 64;   // wave row offset within tile
  const int wc   = (wave >> 1) * 64;  // wave col offset within tile
  const int m0   = blockIdx.y * BM;
  const int n0   = blockIdx.x * BN;

  CF acc[16];
#pragma unroll
  for (int t = 0; t < 16; ++t) acc[t].v = v8f_zero();

  for (int kb = 0; kb < K; kb += BK) {
    // ---- stage A tile: thread t stages row t (K%32==0; row-guarded) ----
    {
      const bool mok = (m0 + tid) < M;
      const float* src = A + (size_t)(m0 + tid) * K + kb;
      if (mok && kb + BK < K) __builtin_prefetch(src + BK, 0, 1); // global_prefetch_b8
      if (mok) {
        const float4* s4 = (const float4*)src;
#pragma unroll
        for (int c = 0; c < 8; ++c) {
          float4 v = s4[c];
          As[tid][c * 4 + 0] = (_Float16)v.x;
          As[tid][c * 4 + 1] = (_Float16)v.y;
          As[tid][c * 4 + 2] = (_Float16)v.z;
          As[tid][c * 4 + 3] = (_Float16)v.w;
        }
      } else {
#pragma unroll
        for (int i = 0; i < BK; ++i) As[tid][i] = (_Float16)0.f;
      }
    }
    // ---- stage B tile transposed into Bs[n][k] ----
    {
      const bool nok = (n0 + tid) < N;
      if (transB) {
        if (nok) {
          const float4* s4 = (const float4*)(B + (size_t)(n0 + tid) * K + kb);
#pragma unroll
          for (int c = 0; c < 8; ++c) {
            float4 v = s4[c];
            Bs[tid][c * 4 + 0] = (_Float16)v.x;
            Bs[tid][c * 4 + 1] = (_Float16)v.y;
            Bs[tid][c * 4 + 2] = (_Float16)v.z;
            Bs[tid][c * 4 + 3] = (_Float16)v.w;
          }
        } else {
#pragma unroll
          for (int i = 0; i < BK; ++i) Bs[tid][i] = (_Float16)0.f;
        }
      } else {
#pragma unroll
        for (int i = 0; i < BK; ++i) {
          float x = nok ? B[(size_t)(kb + i) * N + (n0 + tid)] : 0.f;
          Bs[tid][i] = (_Float16)x;
        }
      }
    }
    __syncthreads();

    // ---- fragments + 16 WMMAs per wave ----
    AF a[4];
#pragma unroll
    for (int rt = 0; rt < 4; ++rt)
#pragma unroll
      for (int p = 0; p < 8; ++p) {
        int k0 = (p >> 2) * 16 + hi * 8 + (p & 3) * 2;     // ISA A layout
        a[rt].u[p] = *(const unsigned*)&As[wr + rt * 16 + ln][k0];
      }
#pragma unroll
    for (int ct = 0; ct < 4; ++ct) {
      AF b;
#pragma unroll
      for (int p = 0; p < 8; ++p) {
        int k0 = (p >> 2) * 16 + hi * 8 + (p & 3) * 2;
        b.u[p] = *(const unsigned*)&Bs[wc + ct * 16 + ln][k0];
      }
#pragma unroll
      for (int rt = 0; rt < 4; ++rt) {
        acc[rt * 4 + ct].v = __builtin_amdgcn_wmma_f32_16x16x32_f16(
            false, a[rt].v, false, b.v, (short)0, acc[rt * 4 + ct].v, false, false);
      }
    }
    __syncthreads();
  }

  // ---- epilogue ----
#pragma unroll
  for (int ct = 0; ct < 4; ++ct) {
    int n = n0 + wc + ct * 16 + ln;
    if (n >= N) continue;
    float bv = bias ? bias[n] : 0.f;
#pragma unroll
    for (int rt = 0; rt < 4; ++rt)
#pragma unroll
      for (int r = 0; r < 8; ++r) {
        int m = m0 + wr + rt * 16 + hi * 8 + r;            // ISA C layout
        if (m >= M) continue;
        float v = acc[rt * 4 + ct].f[r] + bv;
        if (act) v = 0.5f * v * (1.f + erff(v * 0.70710678118654752f)); // exact GELU
        C[(size_t)m * N + n] = v;
      }
  }
}

// =====================================================================
// Fused long-short attention: one wave handles 16 queries of one head.
// Band scores + rel-pos + compressed scores -> softmax -> P@V, all WMMA.
// =====================================================================
__global__ __launch_bounds__(32) void k_attn(
    const float* __restrict__ qw, const float* __restrict__ qr,
    const float* __restrict__ kw, const float* __restrict__ vw,
    const float* __restrict__ kc, const float* __restrict__ vc,
    const float* __restrict__ Pw, const float* __restrict__ Pc,
    float* __restrict__ outp)
{
  __shared__ float sc[16][SCW];

  const int lane = threadIdx.x;
  const int hi   = lane >> 4;
  const int ln   = lane & 15;
  const int bid  = blockIdx.x;
  const int qb   = bid & 127;           // L/16 query blocks
  const int hh   = (bid >> 7) & 15;     // head
  const int b    = bid >> 11;           // batch
  const int l0   = qb * 16;
  const int qc   = l0 >> 6;             // query chunk (S=64)

  for (int i = lane; i < 16 * SCW; i += 32) (&sc[0][0])[i] = 0.f;

  // ---- Q fragments (K = DH = 64 -> two 32-wide k-steps) ----
  AF aqw[2], aqr[2];
  {
    const float* qwp = qw + ((size_t)(b * LL + l0 + ln)) * DD + hh * DHD;
    const float* qrp = qr + ((size_t)(b * LL + l0 + ln)) * DD + hh * DHD;
#pragma unroll
    for (int ks = 0; ks < 2; ++ks) {
#pragma unroll
      for (int i = 0; i < 16; ++i) {
        int kidx = ks * 32 + (i >> 3) * 16 + hi * 8 + (i & 7);
        aqw[ks].h[i] = (_Float16)qwp[kidx];
        aqr[ks].h[i] = (_Float16)qrp[kidx];
      }
    }
  }

  // ---- rel-pos term (qr @ Pw^T), scattered onto band diagonal ----
  for (int w0 = 0; w0 < WW; w0 += 16) {
    CF c; c.v = v8f_zero();
#pragma unroll
    for (int ks = 0; ks < 2; ++ks) {
      AF bf;
#pragma unroll
      for (int i = 0; i < 16; ++i) {
        int kidx = ks * 32 + (i >> 3) * 16 + hi * 8 + (i & 7);
        bf.h[i] = (_Float16)Pw[(size_t)(w0 + ln) * DD + hh * DHD + kidx];
      }
      c.v = __builtin_amdgcn_wmma_f32_16x16x32_f16(
          false, aqr[ks].v, false, bf.v, (short)0, c.v, false, false);
    }
#pragma unroll
    for (int r = 0; r < 8; ++r) {
      int q = r + hi * 8;
      sc[q][q + w0 + ln] += c.f[r];     // j = q + w
    }
  }

  // ---- window score rectangle: qw @ K^T (hw rows l0+1 .. l0+527) ----
  for (int j0 = 0; j0 < 528; j0 += 16) {
    CF c; c.v = v8f_zero();
#pragma unroll
    for (int ks = 0; ks < 2; ++ks) {
      AF bf;
#pragma unroll
      for (int i = 0; i < 16; ++i) {
        int kidx = ks * 32 + (i >> 3) * 16 + hi * 8 + (i & 7);
        int j = j0 + ln;
        float x = 0.f;
        if (j < 527)
          x = kw[((size_t)(b * WLEN + l0 + 1 + j)) * DD + hh * DHD + kidx];
        bf.h[i] = (_Float16)x;
      }
      c.v = __builtin_amdgcn_wmma_f32_16x16x32_f16(
          false, aqw[ks].v, false, bf.v, (short)0, c.v, false, false);
    }
#pragma unroll
    for (int r = 0; r < 8; ++r) {
      int q = r + hi * 8;
      sc[q][j0 + ln] += c.f[r];
    }
  }

  // ---- compressed scores: qw@kc^T + qr@Pc[PIDX]^T ----
  for (int c0 = 0; c0 < NCV; c0 += 16) {
    CF c; c.v = v8f_zero();
#pragma unroll
    for (int ks = 0; ks < 2; ++ks) {
      AF bk, bp;
#pragma unroll
      for (int i = 0; i < 16; ++i) {
        int kidx = ks * 32 + (i >> 3) * 16 + hi * 8 + (i & 7);
        int cc = c0 + ln;
        bk.h[i] = (_Float16)kc[((size_t)(b * NCV + cc)) * DD + hh * DHD + kidx];
        int pidx = 31 - qc + cc;                       // PIDX with R=1
        pidx = pidx < 0 ? 0 : (pidx > 63 ? 63 : pidx);
        bp.h[i] = (_Float16)Pc[(size_t)pidx * DD + hh * DHD + kidx];
      }
      c.v = __builtin_amdgcn_wmma_f32_16x16x32_f16(
          false, aqw[ks].v, false, bk.v, (short)0, c.v, false, false);
      c.v = __builtin_amdgcn_wmma_f32_16x16x32_f16(
          false, aqr[ks].v, false, bp.v, (short)0, c.v, false, false);
    }
#pragma unroll
    for (int r = 0; r < 8; ++r) {
      int q = r + hi * 8;
      sc[q][RECTC + c0 + ln] += c.f[r];
    }
  }

  // ---- masked softmax over 576 valid of 608 cols (single wave) ----
  for (int q = 0; q < 16; ++q) {
    float mx = -1e30f;
    for (int j = lane; j < SCW; j += 32) {
      bool valid;
      if (j < RECTC) { int w = j - q; valid = (j < 527) && (w >= 0) && (w < WW); }
      else           { int cc = j - RECTC; valid = (cc < qc + 32); }  // CMASK
      float v = valid ? sc[q][j] : -1e30f;
      mx = fmaxf(mx, v);
    }
#pragma unroll
    for (int off = 16; off > 0; off >>= 1) mx = fmaxf(mx, __shfl_xor(mx, off));
    float sum = 0.f;
    for (int j = lane; j < SCW; j += 32) {
      bool valid;
      if (j < RECTC) { int w = j - q; valid = (j < 527) && (w >= 0) && (w < WW); }
      else           { int cc = j - RECTC; valid = (cc < qc + 32); }
      float p = valid ? __expf(sc[q][j] - mx) : 0.f;
      sc[q][j] = p;
      sum += p;
    }
#pragma unroll
    for (int off = 16; off > 0; off >>= 1) sum += __shfl_xor(sum, off);
    float inv = 1.f / sum;
    for (int j = lane; j < SCW; j += 32) sc[q][j] *= inv;
  }

  // ---- P @ V  (K = 608, N = DH = 64) ----
  CF o[4];
#pragma unroll
  for (int t = 0; t < 4; ++t) o[t].v = v8f_zero();

  for (int kb = 0; kb < SCW; kb += 32) {
    AF ap;
#pragma unroll
    for (int i = 0; i < 16; ++i) {
      int kidx = kb + (i >> 3) * 16 + hi * 8 + (i & 7);
      ap.h[i] = (_Float16)sc[ln][kidx];
    }
#pragma unroll
    for (int nt = 0; nt < 4; ++nt) {
      AF bv;
#pragma unroll
      for (int i = 0; i < 16; ++i) {
        int kidx = kb + (i >> 3) * 16 + hi * 8 + (i & 7);
        float x = 0.f;
        if (kidx < RECTC) {
          if (kidx < 527)
            x = vw[((size_t)(b * WLEN + l0 + 1 + kidx)) * DD + hh * DHD + nt * 16 + ln];
        } else {
          x = vc[((size_t)(b * NCV + (kidx - RECTC))) * DD + hh * DHD + nt * 16 + ln];
        }
        bv.h[i] = (_Float16)x;
      }
      o[nt].v = __builtin_amdgcn_wmma_f32_16x16x32_f16(
          false, ap.v, false, bv.v, (short)0, o[nt].v, false, false);
    }
  }

#pragma unroll
  for (int nt = 0; nt < 4; ++nt)
#pragma unroll
    for (int r = 0; r < 8; ++r) {
      int q = r + hi * 8;
      outp[((size_t)(b * LL + l0 + q)) * DD + hh * DHD + nt * 16 + ln] = o[nt].f[r];
    }
}

// =====================================================================
// Small elementwise / reduction kernels
// =====================================================================
__global__ __launch_bounds__(256) void k_embed(
    const int* __restrict__ x, const float* __restrict__ emb, float* __restrict__ h)
{
  int row = blockIdx.x;
  int tok = x[row];
  for (int j = threadIdx.x; j < DD; j += 256)
    h[(size_t)row * DD + j] = emb[(size_t)tok * DD + j] * 32.f;   // sqrt(1024)
}

__global__ __launch_bounds__(256) void k_sinpos(float* __restrict__ out, int n)
{
  int r = blockIdx.x;
  float pos = (float)(n - 1 - r);
  for (int j = threadIdx.x; j < DD / 2; j += 256) {
    float inv = powf(10000.f, -(2.f * j) / (float)DD);
    float ang = pos * inv;
    out[(size_t)r * DD + j]          = __sinf(ang);
    out[(size_t)r * DD + DD / 2 + j] = __cosf(ang);
  }
}

// Y = LN(X [+ R]) * g + b, rows of length Dc
__global__ __launch_bounds__(256) void k_layernorm(
    const float* __restrict__ X, const float* __restrict__ R,
    const float* __restrict__ g, const float* __restrict__ bb,
    float* __restrict__ Y, int Dc)
{
  __shared__ float red[256];
  int row = blockIdx.x, t = threadIdx.x;
  const float* x = X + (size_t)row * Dc;
  const float* r = R ? R + (size_t)row * Dc : nullptr;

  float s = 0.f;
  for (int j = t; j < Dc; j += 256) s += x[j] + (r ? r[j] : 0.f);
  red[t] = s; __syncthreads();
  for (int st = 128; st > 0; st >>= 1) { if (t < st) red[t] += red[t + st]; __syncthreads(); }
  float mu = red[0] / Dc; __syncthreads();

  s = 0.f;
  for (int j = t; j < Dc; j += 256) {
    float v = x[j] + (r ? r[j] : 0.f) - mu; s += v * v;
  }
  red[t] = s; __syncthreads();
  for (int st = 128; st > 0; st >>= 1) { if (t < st) red[t] += red[t + st]; __syncthreads(); }
  float inv = rsqrtf(red[0] / Dc + 1e-5f);

  for (int j = t; j < Dc; j += 256) {
    float v = x[j] + (r ? r[j] : 0.f);
    Y[(size_t)row * Dc + j] = (v - mu) * inv * g[j] + bb[j];
  }
}

// chunk softmax compression: block = (b, chunk, head), 64 threads
__global__ __launch_bounds__(64) void k_compress_apply(
    const float* __restrict__ seq, const float* __restrict__ scg,
    float* __restrict__ comp, int len, int nch, int chunk_off)
{
  __shared__ float p[64];
  __shared__ float red[64];
  int bid = blockIdx.x;
  int hh = bid & 15;
  int c  = (bid >> 4) % nch;
  int b  = bid / (16 * nch);
  int t  = threadIdx.x;
  size_t row0 = (size_t)b * len + c * SS;

  float v = scg[(row0 + t) * 16 + hh];
  red[t] = v; __syncthreads();
  for (int s = 32; s > 0; s >>= 1) { if (t < s) red[t] = fmaxf(red[t], red[t + s]); __syncthreads(); }
  float mx = red[0]; __syncthreads();
  float e = __expf(v - mx);
  red[t] = e; __syncthreads();
  for (int s = 32; s > 0; s >>= 1) { if (t < s) red[t] += red[t + s]; __syncthreads(); }
  p[t] = e / red[0];
  __syncthreads();

  float acc = 0.f;
  for (int s = 0; s < SS; ++s)
    acc += p[s] * seq[(row0 + s) * DD + hh * DHD + t];
  comp[((size_t)b * NCV + chunk_off + c) * DD + hh * DHD + t] = acc;
}

// hw = concat(cache[:, -W:], h) — pure row copy; on gfx1250 use the
// CDNA5 async global->LDS->global DMA path (inline asm per ISA ch.10/15.18,
// tracked by ASYNCcnt; low 32 bits of a generic shared pointer are the LDS
// byte address per the aperture rules).
__global__ __launch_bounds__(256) void k_concat_hw(
    const float* __restrict__ cache_l, const float* __restrict__ h, float* __restrict__ hw)
{
  int row = blockIdx.x;               // B * WLEN
  int b = row / WLEN, i = row % WLEN;
  const float* src = (i < WW)
      ? cache_l + ((size_t)b * MM + (MM - WW + i)) * DD
      : h       + ((size_t)b * LL + (i - WW)) * DD;
  float* dst = hw + (size_t)row * DD;
  int t = threadIdx.x;
#if defined(__gfx1250__)
  __shared__ __align__(16) float buf[DD];
  unsigned loff = (unsigned)(size_t)&buf[t * 4];                       // LDS byte addr
  unsigned long long gsrc = (unsigned long long)(size_t)(src + t * 4); // global addr
  asm volatile("global_load_async_to_lds_b128 %0, %1, off"
               :: "v"(loff), "v"(gsrc) : "memory");
  asm volatile("s_wait_asynccnt 0" ::: "memory");
  __syncthreads();
  unsigned long long gdst = (unsigned long long)(size_t)(dst + t * 4);
  asm volatile("global_store_async_from_lds_b128 %0, %1, off"
               :: "v"(gdst), "v"(loff) : "memory");
  asm volatile("s_wait_asynccnt 0" ::: "memory");
#else
  for (int j = t; j < DD; j += 256) dst[j] = src[j];
#endif
}

__global__ __launch_bounds__(256) void k_qadd(
    const float* __restrict__ qt, const float* __restrict__ rw,
    const float* __restrict__ rr, float* __restrict__ qw, float* __restrict__ qr)
{
  size_t row = blockIdx.x;            // B*L
  for (int j = threadIdx.x; j < DD; j += 256) {
    float q = qt[row * DD + j] * 0.125f;   // 1/sqrt(64)
    qw[row * DD + j] = q + rw[j];
    qr[row * DD + j] = q + rr[j];
  }
}

__global__ __launch_bounds__(256) void k_logsoftmax(float* __restrict__ out, int N)
{
  __shared__ float red[256];
  int row = blockIdx.x, t = threadIdx.x;
  float* x = out + (size_t)row * N;
  float mx = -1e30f;
  for (int j = t; j < N; j += 256) mx = fmaxf(mx, x[j]);
  red[t] = mx; __syncthreads();
  for (int st = 128; st > 0; st >>= 1) { if (t < st) red[t] = fmaxf(red[t], red[t + st]); __syncthreads(); }
  mx = red[0]; __syncthreads();
  float s = 0.f;
  for (int j = t; j < N; j += 256) s += __expf(x[j] - mx);
  red[t] = s; __syncthreads();
  for (int st = 128; st > 0; st >>= 1) { if (t < st) red[t] += red[t + st]; __syncthreads(); }
  float lse = mx + __logf(red[0]);
  for (int j = t; j < N; j += 256) x[j] -= lse;
}

// =====================================================================
// Host orchestration
// =====================================================================
static inline void gemm(hipStream_t s, const float* A, const float* B,
                        const float* bias, float* C, int M, int N, int K,
                        int transB, int act)
{
  dim3 grid((N + BN - 1) / BN, (M + BM - 1) / BM);
  k_gemm<<<grid, 128, 0, s>>>(A, B, bias, C, M, N, K, transB, act);
}

extern "C" void kernel_launch(void* const* d_in, const int* in_sizes, int n_in,
                              void* d_out, int out_size, void* d_ws, size_t ws_size,
                              hipStream_t stream)
{
  (void)in_sizes; (void)n_in; (void)out_size; (void)ws_size;

  const int*   x     = (const int*)  d_in[0];
  const float* cache = (const float*)d_in[1];
  const float* emb   = (const float*)d_in[2];
  const float* obias = (const float*)d_in[3];
  const float* Wq    = (const float*)d_in[4];
  const float* Wk    = (const float*)d_in[5];
  const float* Wv    = (const float*)d_in[6];
  const float* Wo    = (const float*)d_in[7];
  const float* wdc   = (const float*)d_in[8];
  const float* bdc   = (const float*)d_in[9];
  const float* Wr    = (const float*)d_in[10];
  const float* Wrc   = (const float*)d_in[11];
  const float* brc   = (const float*)d_in[12];
  const float* rw    = (const float*)d_in[13];
  const float* rr    = (const float*)d_in[14];
  const float* g1    = (const float*)d_in[15];
  const float* b1    = (const float*)d_in[16];
  const float* g2    = (const float*)d_in[17];
  const float* b2    = (const float*)d_in[18];
  const float* gd    = (const float*)d_in[19];
  const float* bd    = (const float*)d_in[20];
  const float* gw    = (const float*)d_in[21];
  const float* bw    = (const float*)d_in[22];
  const float* wf1   = (const float*)d_in[23];
  const float* bf1   = (const float*)d_in[24];
  const float* wf2   = (const float*)d_in[25];
  const float* bf2   = (const float*)d_in[26];
  float* outp = (float*)d_out;

  // ---- workspace carving ----
  float* wsf = (float*)d_ws;
  size_t off = 0;
  auto alloc = [&](size_t n) { float* p = wsf + off; off += n; return p; };

  const size_t ROWS = (size_t)BB * LL;         // 4096
  float* h      = alloc(ROWS * DD);
  float* hw     = alloc((size_t)BB * WLEN * DD);
  float* kw     = alloc((size_t)BB * WLEN * DD);
  float* vw     = alloc((size_t)BB * WLEN * DD);
  float* Xreg   = alloc((size_t)16 * 1024 * 1024);   // 64MB multi-use region
  float* qtmp   = Xreg;                          // dead before FFN
  float* qwb    = Xreg + 4u * 1024 * 1024;
  float* qrb    = Xreg + 8u * 1024 * 1024;
  float* attno  = Xreg + 12u * 1024 * 1024;
  float* ffh    = Xreg;                          // reuses whole region in FFN
  float* ao     = alloc(ROWS * DD);
  float* scbuf  = alloc(ROWS * 16);
  float* comp   = alloc((size_t)BB * NCV * DD);
  float* kc     = alloc((size_t)BB * NCV * DD);
  float* vc     = alloc((size_t)BB * NCV * DD);
  float* poswin = alloc((size_t)WW * DD);
  float* keype  = alloc((size_t)NCV * DD);
  float* Pw     = alloc((size_t)WW * DD);
  float* Pc     = alloc((size_t)NCV * DD);

  // ---- embedding + positional tables ----
  k_embed<<<ROWS, 256, 0, stream>>>(x, emb, h);
  k_sinpos<<<WW, 256, 0, stream>>>(poswin, WW);
  k_sinpos<<<NCV, 256, 0, stream>>>(keype, NCV);

  for (int l = 0; l < NLAY; ++l) {
    const float* cache_l = cache + (size_t)l * BB * MM * DD;
    const float* Wq_l  = Wq  + (size_t)l * DD * DD;
    const float* Wk_l  = Wk  + (size_t)l * DD * DD;
    const float* Wv_l  = Wv  + (size_t)l * DD * DD;
    const float* Wo_l  = Wo  + (size_t)l * DD * DD;
    const float* Wr_l  = Wr  + (size_t)l * DD * DD;
    const float* Wrc_l = Wrc + (size_t)l * DD * DD;
    const float* wdc_l = wdc + (size_t)l * DD * 16;
    const float* bdc_l = bdc + (size_t)l * 16;
    const float* brc_l = brc + (size_t)l * DD;
    const float* rw_l  = rw  + (size_t)l * HH * DHD;
    const float* rr_l  = rr  + (size_t)l * HH * DHD;
    const float* g1_l = g1 + (size_t)l * DD, *b1_l = b1 + (size_t)l * DD;
    const float* g2_l = g2 + (size_t)l * DD, *b2_l = b2 + (size_t)l * DD;
    const float* gd_l = gd + (size_t)l * DD, *bd_l = bd + (size_t)l * DD;
    const float* gw_l = gw + (size_t)l * DD, *bw_l = bw + (size_t)l * DD;
    const float* wf1_l = wf1 + (size_t)l * DD * DFF;
    const float* bf1_l = bf1 + (size_t)l * DFF;
    const float* wf2_l = wf2 + (size_t)l * DFF * DD;
    const float* bf2_l = bf2 + (size_t)l * DD;

    // ---- compression of cache and current sequence ----
    gemm(stream, cache_l, wdc_l, bdc_l, scbuf, BB * MM, 16, DD, 0, 0);
    k_compress_apply<<<BB * 32 * HH, 64, 0, stream>>>(cache_l, scbuf, comp, MM, 32, 0);
    gemm(stream, h, wdc_l, bdc_l, scbuf, BB * LL, 16, DD, 0, 0);
    k_compress_apply<<<BB * 32 * HH, 64, 0, stream>>>(h, scbuf, comp, LL, 32, 32);

    // ---- compressed K/V with dual-LN ----
    gemm(stream, comp, Wk_l, nullptr, kc, BB * NCV, DD, DD, 0, 0);
    k_layernorm<<<BB * NCV, 256, 0, stream>>>(kc, nullptr, gd_l, bd_l, kc, DD);
    gemm(stream, comp, Wv_l, nullptr, vc, BB * NCV, DD, DD, 0, 0);
    k_layernorm<<<BB * NCV, 256, 0, stream>>>(vc, nullptr, gd_l, bd_l, vc, DD);

    // ---- windowed K/V ----
    k_concat_hw<<<BB * WLEN, 256, 0, stream>>>(cache_l, h, hw);
    gemm(stream, hw, Wk_l, nullptr, kw, BB * WLEN, DD, DD, 0, 0);
    k_layernorm<<<BB * WLEN, 256, 0, stream>>>(kw, nullptr, gw_l, bw_l, kw, DD);
    gemm(stream, hw, Wv_l, nullptr, vw, BB * WLEN, DD, DD, 0, 0);
    k_layernorm<<<BB * WLEN, 256, 0, stream>>>(vw, nullptr, gw_l, bw_l, vw, DD);

    // ---- queries (scaled + content/position biases) ----
    gemm(stream, h, Wq_l, nullptr, qtmp, BB * LL, DD, DD, 0, 0);
    k_qadd<<<BB * LL, 256, 0, stream>>>(qtmp, rw_l, rr_l, qwb, qrb);

    // ---- positional projections ----
    gemm(stream, poswin, Wr_l, nullptr, Pw, WW, DD, DD, 0, 0);
    gemm(stream, keype, Wrc_l, brc_l, Pc, NCV, DD, DD, 0, 0);

    // ---- fused attention ----
    k_attn<<<BB * HH * (LL / 16), 32, 0, stream>>>(qwb, qrb, kw, vw, kc, vc, Pw, Pc, attno);

    // ---- output projection + residual LN ----
    gemm(stream, attno, Wo_l, nullptr, ao, BB * LL, DD, DD, 0, 0);
    k_layernorm<<<BB * LL, 256, 0, stream>>>(ao, h, g1_l, b1_l, h, DD);

    // ---- FFN (GELU fused into first GEMM) + residual LN ----
    gemm(stream, h, wf1_l, bf1_l, ffh, BB * LL, DFF, DD, 0, 1);
    gemm(stream, ffh, wf2_l, bf2_l, ao, BB * LL, DD, DFF, 0, 0);
    k_layernorm<<<BB * LL, 256, 0, stream>>>(ao, h, g2_l, b2_l, h, DD);
  }

  // ---- logits = h @ emb^T + bias, then log_softmax in place ----
  gemm(stream, h, emb, obias, outp, BB * LL, VV, DD, 1, 0);
  k_logsoftmax<<<BB * LL, 256, 0, stream>>>(outp, VV);
}
